// SlotAttention_40183714021825
// MI455X (gfx1250) — matrix operand order repaired and checked by reading, not compiled, not run on hardware
//
#include <hip/hip_runtime.h>
#include <stdint.h>

#define N_PTS     100000
#define DIM       64
#define NSEEDS    200
#define SEEDS_PAD 208
#define STILES    13          // 208 / 16 seed tiles
#define MS_ITERS  10
#define EPS2      (0.05f * 0.05f)
#define CHUNK     128         // points per workgroup in the mean-shift kernel
#define MS_WGS    782         // 782*128 = 100096 >= 100000
#define NPAD      (MS_WGS * CHUNK)
#define MS_THREADS (STILES * 32)   // 416 threads = 13 waves (wave32)
#define ARG_WGS   200

typedef __attribute__((ext_vector_type(16))) __bf16 v16bf;
typedef __attribute__((ext_vector_type(8)))  float  v8f;

union BfVec { uint4 q[2]; v16bf v; };

// float -> bf16 bits, round-to-nearest-even (integer-only, always compiles)
__device__ __forceinline__ unsigned short f2bf(float x) {
  unsigned int u = __builtin_bit_cast(unsigned int, x);
  unsigned int r = (u + 0x7FFFu + ((u >> 16) & 1u)) >> 16;
  return (unsigned short)r;
}

// ---------------------------------------------------------------------------
// Init: Z[0]=X[0], Z rest = 0, xnorm = 0.5*||x||^2 (1e30 for pad),
// nearest = ||x - X[0]||
// ---------------------------------------------------------------------------
__global__ void k_init(const float* __restrict__ X, float* __restrict__ Z,
                       float* __restrict__ xnorm, float* __restrict__ nearest) {
  int i = blockIdx.x * blockDim.x + threadIdx.x;
  if (i < SEEDS_PAD * DIM) Z[i] = (i < DIM) ? X[i] : 0.0f;
  if (i < NPAD) {
    if (i < N_PTS) {
      float sq = 0.f, dn = 0.f;
      #pragma unroll 8
      for (int d = 0; d < DIM; ++d) {
        float v = X[i * DIM + d];
        sq += v * v;
        float df = v - X[d];
        dn += df * df;
      }
      xnorm[i]   = 0.5f * sq;
      nearest[i] = sqrtf(dn + 1e-12f);
    } else {
      xnorm[i]   = 1e30f;   // masks padded points out of the softmax
      nearest[i] = -1.0f;
    }
  }
}

// ---------------------------------------------------------------------------
// Farthest-point seeding: two-stage argmax of `nearest`, then distance update
// ---------------------------------------------------------------------------
__global__ void k_argmax_a(const float* __restrict__ nearest,
                           float* __restrict__ pval, int* __restrict__ pidx) {
  __shared__ float sv[256];
  __shared__ int   si[256];
  float best = -2.f; int bi = 0;
  for (int i = blockIdx.x * blockDim.x + threadIdx.x; i < N_PTS;
       i += gridDim.x * blockDim.x) {
    float v = nearest[i];
    if (v > best) { best = v; bi = i; }
  }
  sv[threadIdx.x] = best; si[threadIdx.x] = bi;
  __syncthreads();
  for (int off = 128; off > 0; off >>= 1) {
    if (threadIdx.x < off) {
      float ov = sv[threadIdx.x + off]; int oi = si[threadIdx.x + off];
      if (ov > sv[threadIdx.x] || (ov == sv[threadIdx.x] && oi < si[threadIdx.x])) {
        sv[threadIdx.x] = ov; si[threadIdx.x] = oi;
      }
    }
    __syncthreads();
  }
  if (threadIdx.x == 0) { pval[blockIdx.x] = sv[0]; pidx[blockIdx.x] = si[0]; }
}

__global__ void k_argmax_b(const float* __restrict__ pval,
                           const int* __restrict__ pidx,
                           const float* __restrict__ X,
                           float* __restrict__ Z, int seed) {
  __shared__ float sv[256];
  __shared__ int   si[256];
  int t = threadIdx.x;
  float best = -2.f; int bi = 0;
  if (t < ARG_WGS) { best = pval[t]; bi = pidx[t]; }
  sv[t] = best; si[t] = bi;
  __syncthreads();
  for (int off = 128; off > 0; off >>= 1) {
    if (t < off) {
      float ov = sv[t + off]; int oi = si[t + off];
      if (ov > sv[t] || (ov == sv[t] && oi < si[t])) { sv[t] = ov; si[t] = oi; }
    }
    __syncthreads();
  }
  int idx = si[0];
  if (t < DIM) Z[seed * DIM + t] = X[idx * DIM + t];
}

__global__ void k_dist(const float* __restrict__ X, const float* __restrict__ Z,
                       float* __restrict__ nearest, int seed) {
  int i = blockIdx.x * blockDim.x + threadIdx.x;
  if (i >= N_PTS) return;
  const float* z = Z + seed * DIM;   // uniform -> scalar loads
  float s = 0.f;
  #pragma unroll 8
  for (int d = 0; d < DIM; ++d) { float df = X[i * DIM + d] - z[d]; s += df * df; }
  nearest[i] = fminf(nearest[i], sqrtf(s + 1e-12f));
}

// ---------------------------------------------------------------------------
// Per-iteration prep: Zb = bf16(Z), zn = 0.5*||z||^2, zero num/den
// ---------------------------------------------------------------------------
__global__ void k_prep(const float* __restrict__ Z, unsigned short* __restrict__ Zb,
                       float* __restrict__ zn, float* __restrict__ num,
                       float* __restrict__ den) {
  int i = blockIdx.x * blockDim.x + threadIdx.x;
  if (i < SEEDS_PAD * DIM) { num[i] = 0.f; Zb[i] = f2bf(Z[i]); }
  if (i < SEEDS_PAD) {
    float s = 0.f;
    for (int d = 0; d < DIM; ++d) { float v = Z[i * DIM + d]; s += v * v; }
    zn[i] = 0.5f * s;
    den[i] = 0.f;
  }
}

// ---------------------------------------------------------------------------
// Fused mean-shift iteration (WMMA).
// Per WG: 128-point chunk of X staged in LDS (bf16, row- and dim-major).
// Per wave (13 waves): one 16-seed tile.
//   phase 1: S[16x16] = Z_tile @ X_tile^T  (2x wmma bf16, K=64)
//            w = exp(min(0, S - 0.5||z||^2 - 0.5||x||^2))
//   phase 2: acc[16x80] += W[16x32] @ [X | 1][32x80]  (5x wmma bf16)
//            -> tiles 0..3 are the numerator, tile 4 column 0 is the denom.
// ---------------------------------------------------------------------------
__global__ __launch_bounds__(MS_THREADS) void k_msiter(
    const float* __restrict__ X, const float* __restrict__ xnorm,
    const unsigned short* __restrict__ Zb, const float* __restrict__ zn,
    float* __restrict__ num, float* __restrict__ den) {
  __shared__ __align__(16) unsigned short xrow[CHUNK][DIM];          // point-major
  __shared__ __align__(16) unsigned short xcol[DIM + 16][CHUNK + 8]; // dim-major + ones row
  __shared__ float xns[CHUNK];
  __shared__ __align__(16) unsigned short wbuf[STILES][16][32];      // per-wave W scratch

  const int tid   = threadIdx.x;
  const int gbase = blockIdx.x * CHUNK;

  // cooperative LDS fill: convert f32 -> bf16 once, store both layouts
  for (int idx = tid; idx < CHUNK * DIM; idx += MS_THREADS) {
    int p = idx >> 6, d = idx & 63;
    int g = gbase + p;
    float v = (g < N_PTS) ? X[g * DIM + d] : 0.0f;
    unsigned short b = f2bf(v);
    xrow[p][d] = b;
    xcol[d][p] = b;
  }
  for (int idx = tid; idx < 16 * (CHUNK + 8); idx += MS_THREADS) {
    int r = idx / (CHUNK + 8), c = idx % (CHUNK + 8);
    xcol[DIM + r][c] = (r == 0 && c < CHUNK) ? (unsigned short)0x3F80 : (unsigned short)0;
  }
  for (int p = tid; p < CHUNK; p += MS_THREADS) {
    int g = gbase + p;
    xns[p] = (g < N_PTS) ? xnorm[g] : 1e30f;
  }
  __syncthreads();

  const int wave = tid >> 5;           // 0..12 -> seed tile
  const int lane = tid & 31;
  const int ln   = lane & 15;
  const int half = lane >> 4;
  if (wave >= STILES) return;

  // A operand (Z tile) in WMMA 16-bit A layout: e<8 -> K=k0+half*8+e, e>=8 -> K=k0+16+half*8+(e-8)
  const int row = wave * 16 + ln;
  const unsigned short* zr = Zb + row * DIM;
  BfVec A0, A1;
  A0.q[0] = *(const uint4*)(zr + half * 8);
  A0.q[1] = *(const uint4*)(zr + 16 + half * 8);
  A1.q[0] = *(const uint4*)(zr + 32 + half * 8);
  A1.q[1] = *(const uint4*)(zr + 48 + half * 8);

  float znv[8];
  #pragma unroll
  for (int r = 0; r < 8; ++r) znv[r] = zn[wave * 16 + half * 8 + r];

  v8f acc[5];
  #pragma unroll
  for (int t = 0; t < 5; ++t) acc[t] = (v8f){};

  for (int c32 = 0; c32 < CHUNK / 32; ++c32) {
    // ---- phase 1: scores + exp, two 16-point tiles -> wbuf (bf16, A-layout rows)
    #pragma unroll
    for (int p = 0; p < 2; ++p) {
      const int pt = c32 * 32 + p * 16 + ln;           // point for this lane (B: N=ln)
      const unsigned short* xr = &xrow[pt][0];
      BfVec B0, B1;                                    // B: e<8 -> K = half*8 + e, e>=8 -> K = 16 + half*8 + (e-8)
      B0.q[0] = *(const uint4*)(xr + half * 8);
      B0.q[1] = *(const uint4*)(xr + 16 + half * 8);
      B1.q[0] = *(const uint4*)(xr + 32 + half * 8);
      B1.q[1] = *(const uint4*)(xr + 48 + half * 8);
      v8f s = (v8f){};
      s = __builtin_amdgcn_wmma_f32_16x16x32_bf16(false, A0.v, false, B0.v, (short)0, s, false, false);
      s = __builtin_amdgcn_wmma_f32_16x16x32_bf16(false, A1.v, false, B1.v, (short)0, s, false, false);
      const float xv = xns[pt];
      #pragma unroll
      for (int r = 0; r < 8; ++r) {                    // C: M = half*8 + r, N = ln
        float arg = s[r] - znv[r] - xv;
        float w = __expf(fminf(arg, 0.0f));
        wbuf[wave][half * 8 + r][p * 16 + ln] = f2bf(w);
      }
    }

    // ---- phase 2: A = W (16 seeds x 32 points), B = [X|1] tiles from xcol
    BfVec Aw;
    const unsigned short* wr = &wbuf[wave][ln][0];
    Aw.q[0] = *(const uint4*)(wr + half * 16);
    Aw.q[1] = *(const uint4*)(wr + half * 16 + 8);
    #pragma unroll
    for (int t = 0; t < 5; ++t) {
      const unsigned short* xc = &xcol[t * 16 + ln][c32 * 32 + half * 16];
      BfVec Bt;
      Bt.q[0] = *(const uint4*)(xc);
      Bt.q[1] = *(const uint4*)(xc + 8);
      acc[t] = __builtin_amdgcn_wmma_f32_16x16x32_bf16(false, Aw.v, false, Bt.v, (short)0, acc[t], false, false);
    }
  }

  // ---- spill partial sums (num rows = seeds, cols = dims; tile 4 col 0 = denom)
  #pragma unroll
  for (int t = 0; t < 4; ++t)
    #pragma unroll
    for (int r = 0; r < 8; ++r)
      atomicAdd(&num[(wave * 16 + half * 8 + r) * DIM + t * 16 + ln], acc[t][r]);
  if (ln == 0) {
    #pragma unroll
    for (int r = 0; r < 8; ++r)
      atomicAdd(&den[wave * 16 + half * 8 + r], acc[4][r]);
  }
}

__global__ void k_update(const float* __restrict__ num, const float* __restrict__ den,
                         float* __restrict__ Z) {
  int i = blockIdx.x * blockDim.x + threadIdx.x;
  if (i < NSEEDS * DIM) Z[i] = num[i] / den[i >> 6];
}

// ---------------------------------------------------------------------------
// Sequential connected components over 200 seeds + cluster means (single WG)
// ---------------------------------------------------------------------------
__global__ void k_cc_out(const float* __restrict__ Z, float* __restrict__ out) {
  __shared__ int labels[NSEEDS];
  __shared__ unsigned char comp[NSEEDS];
  __shared__ int cnt[NSEEDS];
  __shared__ int sK, hasLab, curLabel;
  const int t = threadIdx.x;
  if (t < NSEEDS) labels[t] = -1;
  if (t == 0) sK = 0;
  __syncthreads();

  for (int i = 0; i < NSEEDS; ++i) {
    __syncthreads();
    if (labels[i] != -1) continue;       // uniform (shared read)
    if (t == 0) hasLab = 0;
    if (t < NSEEDS) {
      float s = 0.f;
      for (int d = 0; d < DIM; ++d) {
        float df = Z[i * DIM + d] - Z[t * DIM + d];
        s += df * df;
      }
      comp[t] = (s <= EPS2) ? 1 : 0;
    }
    __syncthreads();
    if (t < NSEEDS && comp[t] && labels[t] >= 0) atomicOr(&hasLab, 1);
    if (t < NSEEDS) {
      int c = 0;
      for (int j = 0; j < NSEEDS; ++j) c += (comp[j] && labels[j] == t) ? 1 : 0;
      cnt[t] = c;
    }
    __syncthreads();
    if (t == 0) {
      int mode = 0, bc = cnt[0];
      for (int l = 1; l < NSEEDS; ++l) if (cnt[l] > bc) { bc = cnt[l]; mode = l; }
      if (hasLab) curLabel = mode;
      else { curLabel = sK; sK = sK + 1; }
    }
    __syncthreads();
    if (t < NSEEDS && comp[t]) labels[t] = curLabel;
  }
  __syncthreads();

  for (int idx = t; idx < NSEEDS * DIM; idx += blockDim.x) {
    int l = idx >> 6, d = idx & 63;
    float s = 0.f; int c = 0;
    for (int j = 0; j < NSEEDS; ++j)
      if (labels[j] == l) { s += Z[j * DIM + d]; c++; }
    out[idx] = s / ((float)c + 1e-8f);
  }
}

// ---------------------------------------------------------------------------
extern "C" void kernel_launch(void* const* d_in, const int* in_sizes, int n_in,
                              void* d_out, int out_size, void* d_ws, size_t ws_size,
                              hipStream_t stream) {
  const float* X = (const float*)d_in[0];
  float* out = (float*)d_out;

  char* ws = (char*)d_ws;
  size_t off = 0;
  auto alloc = [&](size_t bytes) -> void* {
    off = (off + 255) & ~(size_t)255;
    void* p = ws + off;
    off += bytes;
    return p;
  };
  float*          Z       = (float*)          alloc(SEEDS_PAD * DIM * sizeof(float));
  unsigned short* Zb      = (unsigned short*) alloc(SEEDS_PAD * DIM * sizeof(unsigned short));
  float*          zn      = (float*)          alloc(SEEDS_PAD * sizeof(float));
  float*          num     = (float*)          alloc(SEEDS_PAD * DIM * sizeof(float));
  float*          den     = (float*)          alloc(SEEDS_PAD * sizeof(float));
  float*          xnorm   = (float*)          alloc(NPAD * sizeof(float));
  float*          nearest = (float*)          alloc(NPAD * sizeof(float));
  float*          pval    = (float*)          alloc(ARG_WGS * sizeof(float));
  int*            pidx    = (int*)            alloc(ARG_WGS * sizeof(int));
  (void)ws_size; (void)in_sizes; (void)n_in; (void)out_size;

  // --- seeding ---
  k_init<<<NPAD / 256, 256, 0, stream>>>(X, Z, xnorm, nearest);
  for (int s = 1; s < NSEEDS; ++s) {
    k_argmax_a<<<ARG_WGS, 256, 0, stream>>>(nearest, pval, pidx);
    k_argmax_b<<<1, 256, 0, stream>>>(pval, pidx, X, Z, s);
    if (s < NSEEDS - 1)
      k_dist<<<(N_PTS + 255) / 256, 256, 0, stream>>>(X, Z, nearest, s);
  }

  // --- hill climb (WMMA) ---
  for (int it = 0; it < MS_ITERS; ++it) {
    k_prep<<<(SEEDS_PAD * DIM + 255) / 256, 256, 0, stream>>>(Z, Zb, zn, num, den);
    k_msiter<<<MS_WGS, MS_THREADS, 0, stream>>>(X, xnorm, Zb, zn, num, den);
    k_update<<<(NSEEDS * DIM + 255) / 256, 256, 0, stream>>>(num, den, Z);
  }

  // --- connected components + means ---
  k_cc_out<<<1, 256, 0, stream>>>(Z, out);
}
